// GraphModel_2473901162945
// MI455X (gfx1250) — matrix operand order repaired and checked by reading, not compiled, not run on hardware
//
#include <hip/hip_runtime.h>
#include <hip/hip_bf16.h>
#include <math.h>

// ---------------------------------------------------------------------------
// Problem constants (from reference)
// ---------------------------------------------------------------------------
#define NNODES 12000
#define NEDGES 384000
#define FIN    512
#define FHID   256
#define FOUT   64

typedef __attribute__((ext_vector_type(16))) __bf16 v16bf;
typedef __attribute__((ext_vector_type(8)))  float  v8f;

union FragU {
    uint4 u[2];   // 32 bytes
    v16bf v;      // 16 x bf16 = 32 bytes
};

static __device__ __forceinline__ unsigned short f2bf(float f) {
    // round-to-nearest-even f32 -> bf16
    unsigned u = __float_as_uint(f);
    unsigned r = (u + 0x7FFFu + ((u >> 16) & 1u)) >> 16;
    return (unsigned short)r;
}

// ---------------------------------------------------------------------------
// Utility kernels
// ---------------------------------------------------------------------------
__global__ void k_fill_zero(float* p, int n) {
    int i = blockIdx.x * blockDim.x + threadIdx.x;
    if (i < n) p[i] = 0.0f;
}

__global__ void k_deg(const int* col, float* deg, int e) {
    int i = blockIdx.x * blockDim.x + threadIdx.x;
    if (i < e) atomicAdd(&deg[col[i]], 1.0f);
}

// deg -> dinv in place; +1.0 accounts for the self-loop, so deg+1 >= 1 always
__global__ void k_dinv(float* deg, int n) {
    int i = blockIdx.x * blockDim.x + threadIdx.x;
    if (i < n) deg[i] = rsqrtf(deg[i] + 1.0f);
}

__global__ void k_cvt(const float* in, unsigned short* out, int n) {
    int i = blockIdx.x * blockDim.x + threadIdx.x;
    if (i < n) out[i] = f2bf(in[i]);
}

// W (K x N row-major, f32) -> Wt (N x K row-major, bf16)
__global__ void k_cvt_t(const float* W, unsigned short* Wt, int K, int N) {
    int i = blockIdx.x * blockDim.x + threadIdx.x;
    if (i < K * N) {
        int k = i / N, n = i % N;
        Wt[(size_t)n * K + k] = f2bf(W[i]);
    }
}

// out[col] += (dinv[row]*dinv[col]) * hw[row]  over E real edges + N self-loops
__global__ void k_spmm(const float* __restrict__ hw, const int* __restrict__ erow,
                       const int* __restrict__ ecol, const float* __restrict__ dinv,
                       float* __restrict__ out, int E, int n, int F) {
    int F4 = F >> 2;
    int tid = blockIdx.x * blockDim.x + threadIdx.x;
    int total = (E + n) * F4;
    if (tid >= total) return;
    int e = tid / F4;
    int f = (tid % F4) << 2;
    int r, c;
    if (e < E) { r = erow[e]; c = ecol[e]; }
    else       { r = e - E;  c = r; }
    float nrm = dinv[r] * dinv[c];
    float4 v = *(const float4*)(hw + (size_t)r * F + f);
    float* o = out + (size_t)c * F + f;
    atomicAdd(o + 0, nrm * v.x);
    atomicAdd(o + 1, nrm * v.y);
    atomicAdd(o + 2, nrm * v.z);
    atomicAdd(o + 3, nrm * v.w);
}

// h = act(agg + bias) -> bf16 (only bf16 needed downstream)
__global__ void k_bias_act(const float* agg, const float* bias,
                           unsigned short* outb, int total, int F, int do_relu) {
    int i = blockIdx.x * blockDim.x + threadIdx.x;
    if (i >= total) return;
    float v = agg[i] + bias[i % F];
    if (do_relu) v = fmaxf(v, 0.0f);
    outb[i] = f2bf(v);
}

// ---------------------------------------------------------------------------
// WMMA GEMM:  C[M,N] = A[M,K] * Bt[N,K]^T   (bf16 in, f32 accumulate)
// One wave computes a 32x32 tile as 2x2 of 16x16 WMMAs. Requires
// M%32==0, N%32==0, K%32==0 (true for all call sites here).
// do_sigmoid: apply 1/(1+exp(-x)) in the epilogue (final decoder GEMM).
// ---------------------------------------------------------------------------
__global__ void k_gemm_wmma(const unsigned short* __restrict__ A,
                            const unsigned short* __restrict__ Bt,
                            float* __restrict__ C,
                            int M, int N, int K, int do_sigmoid) {
    int gtid = blockIdx.x * blockDim.x + threadIdx.x;
    int wave = gtid >> 5;
    int lane = threadIdx.x & 31;
    int ntiles = N >> 5;
    int mtiles = M >> 5;
    if (wave >= mtiles * ntiles) return;           // wave-uniform: EXEC all-1s inside
    int mt = (wave / ntiles) << 5;
    int nt = (wave % ntiles) << 5;

    int lrow = lane & 15;     // M (for A) / N (for B) index within tile
    int half = lane >> 4;

    v8f acc00 = {}, acc01 = {}, acc10 = {}, acc11 = {};

    for (int k = 0; k < K; k += 32) {
        FragU a0, a1, b0, b1;
        // A fragments: per lane K = 16*half + 0..15 contiguous
        const unsigned short* ap0 = A + (size_t)(mt + lrow) * K + k;
        const unsigned short* ap1 = A + (size_t)(mt + 16 + lrow) * K + k;
        a0.u[0] = *(const uint4*)(ap0 + 16 * half);
        a0.u[1] = *(const uint4*)(ap0 + 16 * half + 8);
        a1.u[0] = *(const uint4*)(ap1 + 16 * half);
        a1.u[1] = *(const uint4*)(ap1 + 16 * half + 8);
        // B fragments (from transposed operand): per lane K = 16*half + 0..15 contiguous
        const unsigned short* bp0 = Bt + (size_t)(nt + lrow) * K + k + 16 * half;
        const unsigned short* bp1 = Bt + (size_t)(nt + 16 + lrow) * K + k + 16 * half;
        b0.u[0] = *(const uint4*)(bp0);
        b0.u[1] = *(const uint4*)(bp0 + 8);
        b1.u[0] = *(const uint4*)(bp1);
        b1.u[1] = *(const uint4*)(bp1 + 8);

        acc00 = __builtin_amdgcn_wmma_f32_16x16x32_bf16(false, a0.v, false, b0.v,
                                                        (short)0, acc00, false, false);
        acc01 = __builtin_amdgcn_wmma_f32_16x16x32_bf16(false, a0.v, false, b1.v,
                                                        (short)0, acc01, false, false);
        acc10 = __builtin_amdgcn_wmma_f32_16x16x32_bf16(false, a1.v, false, b0.v,
                                                        (short)0, acc10, false, false);
        acc11 = __builtin_amdgcn_wmma_f32_16x16x32_bf16(false, a1.v, false, b1.v,
                                                        (short)0, acc11, false, false);
    }

    // C/D layout: VGPR r holds row (r + 8*half), col = lane&15
    int ccol = lrow;
    for (int r = 0; r < 8; ++r) {
        int rr = r + 8 * half;
        float v00 = acc00[r], v01 = acc01[r], v10 = acc10[r], v11 = acc11[r];
        if (do_sigmoid) {
            v00 = 1.0f / (1.0f + expf(-v00));
            v01 = 1.0f / (1.0f + expf(-v01));
            v10 = 1.0f / (1.0f + expf(-v10));
            v11 = 1.0f / (1.0f + expf(-v11));
        }
        C[(size_t)(mt + rr) * N + (nt + ccol)]           = v00;
        C[(size_t)(mt + rr) * N + (nt + 16 + ccol)]      = v01;
        C[(size_t)(mt + 16 + rr) * N + (nt + ccol)]      = v10;
        C[(size_t)(mt + 16 + rr) * N + (nt + 16 + ccol)] = v11;
    }
}

// ---------------------------------------------------------------------------
// Host-side orchestration
// ---------------------------------------------------------------------------
static inline size_t alignup(size_t x) { return (x + 255) & ~(size_t)255; }
static inline int nblk(long long n, int b) { return (int)((n + b - 1) / b); }

extern "C" void kernel_launch(void* const* d_in, const int* in_sizes, int n_in,
                              void* d_out, int out_size, void* d_ws, size_t ws_size,
                              hipStream_t stream) {
    (void)in_sizes; (void)n_in; (void)out_size; (void)ws_size;
    const float* x    = (const float*)d_in[0];
    const int*   ei   = (const int*)d_in[1];   // [2,E]: row = ei, col = ei+E
    const float* W1   = (const float*)d_in[2];
    const float* b1   = (const float*)d_in[3];
    const float* W2   = (const float*)d_in[4];
    const float* b2   = (const float*)d_in[5];
    const float* Wmu  = (const float*)d_in[6];
    const float* bmu  = (const float*)d_in[7];
    // d_in[8]=Wls, d_in[9]=bls: dead for the output (z = mu) -> skipped
    float* out = (float*)d_out;

    const int Nn = NNODES, E = NEDGES;
    const int* erow = ei;
    const int* ecol = ei + E;

    // Workspace carve-up
    char* w = (char*)d_ws;
    float*          deg = (float*)w;          w += alignup((size_t)Nn * 4);
    unsigned short* Xb  = (unsigned short*)w; w += alignup((size_t)Nn * FIN * 2);
    unsigned short* W1t = (unsigned short*)w; w += alignup((size_t)FHID * FIN * 2);
    unsigned short* W2t = (unsigned short*)w; w += alignup((size_t)FHID * FHID * 2);
    unsigned short* Wmt = (unsigned short*)w; w += alignup((size_t)FOUT * FHID * 2);
    float*          HW  = (float*)w;          w += alignup((size_t)Nn * FHID * 4);
    float*          AGG = (float*)w;          w += alignup((size_t)Nn * FHID * 4);
    unsigned short* Hb  = (unsigned short*)w; w += alignup((size_t)Nn * FHID * 2);
    unsigned short* Zb  = (unsigned short*)w; w += alignup((size_t)Nn * FOUT * 2);

    const int B = 256;

    // 1) normalization: deg (with self-loop) -> dinv (in place)
    k_fill_zero<<<nblk(Nn, B), B, 0, stream>>>(deg, Nn);
    k_deg<<<nblk(E, B), B, 0, stream>>>(ecol, deg, E);
    k_dinv<<<nblk(Nn, B), B, 0, stream>>>(deg, Nn);

    // 2) operand conversions
    k_cvt<<<nblk((long long)Nn * FIN, B), B, 0, stream>>>(x, Xb, Nn * FIN);
    k_cvt_t<<<nblk(FIN * FHID, B), B, 0, stream>>>(W1, W1t, FIN, FHID);
    k_cvt_t<<<nblk(FHID * FHID, B), B, 0, stream>>>(W2, W2t, FHID, FHID);
    k_cvt_t<<<nblk(FHID * FOUT, B), B, 0, stream>>>(Wmu, Wmt, FHID, FOUT);

    // 3) layer 1: HW = X @ W1 ; AGG = scatter(norm * HW[row]) ; Hb = relu(AGG + b1)
    {
        long long waves = (long long)(Nn / 32) * (FHID / 32);
        k_gemm_wmma<<<nblk(waves * 32, B), B, 0, stream>>>(Xb, W1t, HW, Nn, FHID, FIN, 0);
        k_fill_zero<<<nblk((long long)Nn * FHID, B), B, 0, stream>>>(AGG, Nn * FHID);
        k_spmm<<<nblk((long long)(E + Nn) * (FHID / 4), B), B, 0, stream>>>(
            HW, erow, ecol, deg, AGG, E, Nn, FHID);
        k_bias_act<<<nblk((long long)Nn * FHID, B), B, 0, stream>>>(
            AGG, b1, Hb, Nn * FHID, FHID, 1);
    }

    // 4) layer 2
    {
        long long waves = (long long)(Nn / 32) * (FHID / 32);
        k_gemm_wmma<<<nblk(waves * 32, B), B, 0, stream>>>(Hb, W2t, HW, Nn, FHID, FHID, 0);
        k_fill_zero<<<nblk((long long)Nn * FHID, B), B, 0, stream>>>(AGG, Nn * FHID);
        k_spmm<<<nblk((long long)(E + Nn) * (FHID / 4), B), B, 0, stream>>>(
            HW, erow, ecol, deg, AGG, E, Nn, FHID);
        k_bias_act<<<nblk((long long)Nn * FHID, B), B, 0, stream>>>(
            AGG, b2, Hb, Nn * FHID, FHID, 1);
    }

    // 5) mu layer -> z (bf16), no relu
    {
        long long waves = (long long)(Nn / 32) * (FOUT / 32);
        k_gemm_wmma<<<nblk(waves * 32, B), B, 0, stream>>>(Hb, Wmt, HW, Nn, FOUT, FHID, 0);
        k_fill_zero<<<nblk((long long)Nn * FOUT, B), B, 0, stream>>>(AGG, Nn * FOUT);
        k_spmm<<<nblk((long long)(E + Nn) * (FOUT / 4), B), B, 0, stream>>>(
            HW, erow, ecol, deg, AGG, E, Nn, FOUT);
        k_bias_act<<<nblk((long long)Nn * FOUT, B), B, 0, stream>>>(
            AGG, bmu, Zb, Nn * FOUT, FOUT, 0);
    }

    // 6) decoder: out = sigmoid(Z @ Z^T)  -- B operand is Z itself (already "transposed")
    {
        long long waves = (long long)(Nn / 32) * (Nn / 32);
        k_gemm_wmma<<<nblk(waves * 32, B), B, 0, stream>>>(Zb, Zb, out, Nn, Nn, FOUT, 1);
    }
}
